// AutoregressiveRNN_38792144617754
// MI455X (gfx1250) — matrix operand order repaired and checked by reading, not compiled, not run on hardware
//
#include <hip/hip_runtime.h>
#include <hip/hip_bf16.h>
#include <math.h>

// Problem sizes (fixed by the reference)
#define B_    256
#define S_    512
#define V_    32
#define E_    128
#define H_    512
#define FOURH 2048
#define MROWS (B_ * S_)   // 131072

typedef __attribute__((ext_vector_type(16))) __bf16 v16bf;
typedef __attribute__((ext_vector_type(8)))  __bf16 v8bf;
typedef __attribute__((ext_vector_type(8)))  float  v8f;

static __device__ inline v8f vzero8f() {
  v8f z;
#pragma unroll
  for (int i = 0; i < 8; ++i) z[i] = 0.f;
  return z;
}

// Build a 16-elem bf16 fragment from two contiguous 16B LDS chunks.
static __device__ inline v16bf ld_frag16(const __bf16* lo, const __bf16* hi) {
  v8bf l = *reinterpret_cast<const v8bf*>(lo);
  v8bf h = *reinterpret_cast<const v8bf*>(hi);
  return __builtin_shufflevector(l, h, 0,1,2,3,4,5,6,7,8,9,10,11,12,13,14,15);
}

static __device__ inline float sigmoidf_(float x) { return 1.f / (1.f + __expf(-x)); }

// LDS byte offset of a __shared__ object (generic ptr low 32 bits = group-seg offset).
static __device__ inline unsigned lds_off(const void* p) {
  return (unsigned)(unsigned long long)p;
}

// gfx1250 async DMA: LDS[vdst] = MEM[vaddr], 16B per lane, tracked by ASYNCcnt.
static __device__ inline void async_ld_b128(unsigned lds, const void* gaddr) {
  asm volatile("global_load_async_to_lds_b128 %0, %1, off"
               :: "v"(lds), "v"(gaddr) : "memory");
}
static __device__ inline void wait_async_le4() {
  asm volatile("s_wait_asynccnt 0x4" ::: "memory");
}
static __device__ inline void wait_async_0() {
  asm volatile("s_wait_asynccnt 0x0" ::: "memory");
}

// ---------------------------------------------------------------------------
// Prep kernels
// ---------------------------------------------------------------------------

// out[n*K + k] = (bf16) in[k*N + n]   (weights [K,N] f32 -> [N,K] bf16)
__global__ void transpose_to_bf16(const float* __restrict__ in, __bf16* __restrict__ out,
                                  int K, int N) {
  long idx = (long)blockIdx.x * blockDim.x + threadIdx.x;
  long total = (long)K * N;
  if (idx >= total) return;
  int n = (int)(idx / K);
  int k = (int)(idx - (long)n * K);
  out[idx] = (__bf16)in[(long)k * N + n];
}

// x[b,s,e] = emb[tok[b,s], e] * sqrt(E), bf16
__global__ void embed_kernel(const int* __restrict__ tokens, const float* __restrict__ emb,
                             __bf16* __restrict__ x) {
  long idx = (long)blockIdx.x * blockDim.x + threadIdx.x;
  if (idx >= (long)MROWS * E_) return;
  int e  = (int)(idx & (E_ - 1));
  long bs = idx >> 7;              // E_ = 128
  int tok = tokens[bs];
  x[idx] = (__bf16)(emb[(long)tok * E_ + e] * 11.313708499f); // sqrt(128)
}

__global__ void fill_zero_u32(unsigned int* __restrict__ p, int nwords) {
  int i = blockIdx.x * blockDim.x + threadIdx.x;
  if (i < nwords) p[i] = 0u;
}

// ---------------------------------------------------------------------------
// Generic WMMA GEMM:  out[M,N] = act(A[M,K] * Wt^T + bias)
//   A: bf16 row-major [M,K]; Wt: bf16 N-major [N,K]; bias: f32 [N] or null.
//   Block tile 128x128, 8 waves, wave tile 64x32 (4x2 16x16 WMMA C).
//   Double-buffered LDS fed by global_load_async_to_lds_b128 (ASYNCcnt).
//   M % 128 == 0 and K % 32 == 0 (true for all calls); N bounds-checked.
//   Out-of-range B rows leave garbage in LDS; they only touch C columns >= N
//   which are never stored, so no zero-fill is required.
// ---------------------------------------------------------------------------
__global__ __launch_bounds__(256)
void wmma_gemm_bias(const __bf16* __restrict__ A, const __bf16* __restrict__ Wt,
                    const float* __restrict__ bias,
                    float* __restrict__ outF, __bf16* __restrict__ outB,
                    int M, int N, int K, int relu) {
  __shared__ __attribute__((aligned(16))) __bf16 As[2][128][40];
  __shared__ __attribute__((aligned(16))) __bf16 Bs[2][128][40];

  const int tid  = threadIdx.x;
  const int lane = tid & 31;
  const int wid  = tid >> 5;
  const int wm   = (wid & 1) * 64;   // wave row offset in block tile
  const int wn   = (wid >> 1) * 32;  // wave col offset in block tile
  const long m0  = (long)blockIdx.x * 128;
  const int  n0  = blockIdx.y * 128;

  v8f acc[4][2];
#pragma unroll
  for (int mi = 0; mi < 4; ++mi)
#pragma unroll
    for (int ni = 0; ni < 2; ++ni) acc[mi][ni] = vzero8f();

  const int rbase = tid >> 2;        // 0..63
  const int kc    = (tid & 3) * 8;   // 0,8,16,24

  // issue 4 async b128 DMAs for one 128x32 A tile + 128x32 B tile
  auto stage = [&](int k0, int sel) {
#pragma unroll
    for (int i = 0; i < 2; ++i) {
      int r = rbase + i * 64;
      async_ld_b128(lds_off(&As[sel][r][kc]), &A[(m0 + r) * K + k0 + kc]);
      int ng = n0 + r;
      if (ng < N)
        async_ld_b128(lds_off(&Bs[sel][r][kc]), &Wt[(long)ng * K + k0 + kc]);
    }
  };

  const int nk = K >> 5;
  stage(0, 0);

  for (int kk = 0; kk < nk; ++kk) {
    const int sel = kk & 1;
    if (kk + 1 < nk) {
      stage((kk + 1) << 5, sel ^ 1);   // prefetch next tile into other buffer
      wait_async_le4();                // in-order completion => current tile landed
    } else {
      wait_async_0();
    }
    __syncthreads();

    const int kh = lane >> 4;
    v16bf af[4];
#pragma unroll
    for (int mi = 0; mi < 4; ++mi) {
      int ar = wm + mi * 16 + (lane & 15);
      af[mi] = ld_frag16(&As[sel][ar][kh * 8], &As[sel][ar][16 + kh * 8]);
    }
#pragma unroll
    for (int ni = 0; ni < 2; ++ni) {
      int bc = wn + ni * 16 + (lane & 15);
      v16bf bfr = ld_frag16(&Bs[sel][bc][kh * 8], &Bs[sel][bc][16 + kh * 8]);
#pragma unroll
      for (int mi = 0; mi < 4; ++mi)
        acc[mi][ni] = __builtin_amdgcn_wmma_f32_16x16x32_bf16(
            false, af[mi], false, bfr, (short)0, acc[mi][ni], false, false);
    }
    __syncthreads();   // all reads of buffer `sel` done before it is re-filled
  }

  const int rof = (lane >> 4) * 8;
  const int cn  = lane & 15;
#pragma unroll
  for (int mi = 0; mi < 4; ++mi)
#pragma unroll
    for (int ni = 0; ni < 2; ++ni) {
      int gn = n0 + wn + ni * 16 + cn;
      if (gn >= N) continue;
      float bsum = bias ? bias[gn] : 0.f;
#pragma unroll
      for (int j = 0; j < 8; ++j) {
        long gm = m0 + wm + mi * 16 + j + rof;
        float v = acc[mi][ni][j] + bsum;
        if (relu) v = v > 0.f ? v : 0.f;
        if (outF) outF[gm * N + gn] = v;
        else      outB[gm * N + gn] = (__bf16)v;
      }
    }
}

// ---------------------------------------------------------------------------
// Fused LSTM step:  z = xz[:,t,:] + h_{t-1} @ U ; gates ; c,h update.
// Grid (B/128, H/32), 256 threads (8 waves). Each block owns 128 batch rows x
// 32 h-cols across ALL FOUR gates so the nonlinearities fuse locally.
// Same async double-buffered LDS pipeline as the GEMM.
// h is double-buffered across steps (h_in read-only, h_out write-only).
// ---------------------------------------------------------------------------
__global__ __launch_bounds__(256)
void lstm_step(const float* __restrict__ xz, const __bf16* __restrict__ h_in,
               const __bf16* __restrict__ Ut,   // [4H, H] N-major bf16
               float* __restrict__ c, __bf16* __restrict__ h_out,
               __bf16* __restrict__ x_out, int t) {
  __shared__ __attribute__((aligned(16))) __bf16 Hs[2][128][40];
  __shared__ __attribute__((aligned(16))) __bf16 Us[2][128][40];

  const int tid  = threadIdx.x;
  const int lane = tid & 31;
  const int wid  = tid >> 5;
  const int bb0  = blockIdx.x * 128;   // batch rows
  const int n0   = blockIdx.y * 32;    // h cols (within each gate)

  v8f acc[4][2];
#pragma unroll
  for (int g = 0; g < 4; ++g)
#pragma unroll
    for (int ccx = 0; ccx < 2; ++ccx) acc[g][ccx] = vzero8f();

  const int rbase = tid >> 2;
  const int kc    = (tid & 3) * 8;

  auto stage = [&](int k0, int sel) {
#pragma unroll
    for (int i = 0; i < 2; ++i) {
      int r = rbase + i * 64;
      async_ld_b128(lds_off(&Hs[sel][r][kc]),
                    &h_in[(long)(bb0 + r) * H_ + k0 + kc]);
      int g = r >> 5, ncl = r & 31;    // 128 "n rows" = 4 gates x 32 cols
      async_ld_b128(lds_off(&Us[sel][r][kc]),
                    &Ut[(long)(g * H_ + n0 + ncl) * H_ + k0 + kc]);
    }
  };

  const int nk = H_ >> 5;   // 16
  stage(0, 0);

  for (int kk = 0; kk < nk; ++kk) {
    const int sel = kk & 1;
    if (kk + 1 < nk) {
      stage((kk + 1) << 5, sel ^ 1);
      wait_async_le4();
    } else {
      wait_async_0();
    }
    __syncthreads();

    const int kh = lane >> 4;
    int ar = wid * 16 + (lane & 15);
    v16bf af = ld_frag16(&Hs[sel][ar][kh * 8], &Hs[sel][ar][16 + kh * 8]);
#pragma unroll
    for (int g = 0; g < 4; ++g)
#pragma unroll
      for (int ccx = 0; ccx < 2; ++ccx) {
        int bc = g * 32 + ccx * 16 + (lane & 15);
        v16bf bfr = ld_frag16(&Us[sel][bc][kh * 8], &Us[sel][bc][16 + kh * 8]);
        acc[g][ccx] = __builtin_amdgcn_wmma_f32_16x16x32_bf16(
            false, af, false, bfr, (short)0, acc[g][ccx], false, false);
      }
    __syncthreads();
  }

  const int rof = (lane >> 4) * 8;
  const int cn  = lane & 15;
#pragma unroll
  for (int ccx = 0; ccx < 2; ++ccx) {
    int hcol = n0 + ccx * 16 + cn;
#pragma unroll
    for (int j = 0; j < 8; ++j) {
      int b   = bb0 + wid * 16 + j + rof;
      long zr = ((long)b * S_ + t) * (long)FOURH;
      float zi = acc[0][ccx][j] + xz[zr + 0 * H_ + hcol];
      float zf = acc[1][ccx][j] + xz[zr + 1 * H_ + hcol];
      float zg = acc[2][ccx][j] + xz[zr + 2 * H_ + hcol];
      float zo = acc[3][ccx][j] + xz[zr + 3 * H_ + hcol];
      float iv = sigmoidf_(zi);
      float fv = sigmoidf_(zf);
      float gv = tanhf(zg);
      float ov = sigmoidf_(zo);
      long  ci = (long)b * H_ + hcol;
      float cv = fv * c[ci] + iv * gv;
      c[ci] = cv;
      float hv = ov * tanhf(cv);
      __bf16 hb = (__bf16)hv;
      h_out[ci] = hb;
      x_out[((long)b * S_ + t) * H_ + hcol] = hb;
    }
  }
}

// ---------------------------------------------------------------------------
// Host orchestration
// ---------------------------------------------------------------------------
extern "C" void kernel_launch(void* const* d_in, const int* in_sizes, int n_in,
                              void* d_out, int out_size, void* d_ws, size_t ws_size,
                              hipStream_t stream) {
  const int*   tokens = (const int*)d_in[0];
  const float* emb    = (const float*)d_in[1];
  const float* w0     = (const float*)d_in[2];
  const float* u0     = (const float*)d_in[3];
  const float* b0     = (const float*)d_in[4];
  const float* w_rest = (const float*)d_in[5];
  const float* u_rest = (const float*)d_in[6];
  const float* b_rest = (const float*)d_in[7];
  const float* dw1    = (const float*)d_in[8];
  const float* db1    = (const float*)d_in[9];
  const float* dw2    = (const float*)d_in[10];
  const float* db2    = (const float*)d_in[11];
  float* out = (float*)d_out;

  // bump allocator over workspace (256B aligned)
  char* ws = (char*)d_ws;
  size_t off = 0;
  auto alloc = [&](size_t bytes) -> char* {
    char* p = ws + off;
    off = (off + bytes + 255) & ~(size_t)255;
    return p;
  };

  __bf16* xA    = (__bf16*)alloc((size_t)MROWS * H_ * 2);      // ping (layer I/O)
  __bf16* xB    = (__bf16*)alloc((size_t)MROWS * H_ * 2);      // pong
  float*  xz    = (float*) alloc((size_t)MROWS * FOURH * 4);   // input projections
  __bf16* h1    = (__bf16*)xz;  // dense1 output reuses xz region (free by then)
  __bf16* hping = (__bf16*)alloc((size_t)B_ * H_ * 2);
  __bf16* hpong = (__bf16*)alloc((size_t)B_ * H_ * 2);
  float*  cbuf  = (float*) alloc((size_t)B_ * H_ * 4);
  __bf16* w0t   = (__bf16*)alloc((size_t)FOURH * E_ * 2);
  __bf16* u0t   = (__bf16*)alloc((size_t)FOURH * H_ * 2);
  __bf16* wrt   = (__bf16*)alloc((size_t)3 * FOURH * H_ * 2);
  __bf16* urt   = (__bf16*)alloc((size_t)3 * FOURH * H_ * 2);
  __bf16* dw1t  = (__bf16*)alloc((size_t)1024 * 512 * 2);
  __bf16* dw2t  = (__bf16*)alloc((size_t)32 * 1024 * 2);

  auto trans = [&](const float* in, __bf16* o, int K, int N) {
    long total = (long)K * N;
    transpose_to_bf16<<<(int)((total + 255) / 256), 256, 0, stream>>>(in, o, K, N);
  };

  // Weight prep (one-time cost, amortized over graph replays)
  trans(w0, w0t, E_, FOURH);
  trans(u0, u0t, H_, FOURH);
  for (int l = 0; l < 3; ++l) {
    trans(w_rest + (size_t)l * H_ * FOURH, wrt + (size_t)l * FOURH * H_, H_, FOURH);
    trans(u_rest + (size_t)l * H_ * FOURH, urt + (size_t)l * FOURH * H_, H_, FOURH);
  }
  trans(dw1, dw1t, H_, 2 * H_);
  trans(dw2, dw2t, 2 * H_, V_);

  // Embedding
  {
    long total = (long)MROWS * E_;
    embed_kernel<<<(int)((total + 255) / 256), 256, 0, stream>>>(tokens, emb, xA);
  }

  __bf16* xcur  = xA;
  __bf16* xnext = xB;
  for (int l = 0; l < 4; ++l) {
    const int     Kin  = (l == 0) ? E_ : H_;
    const __bf16* Wt   = (l == 0) ? w0t : wrt + (size_t)(l - 1) * FOURH * H_;
    const __bf16* Ut   = (l == 0) ? u0t : urt + (size_t)(l - 1) * FOURH * H_;
    const float*  bias = (l == 0) ? b0  : b_rest + (size_t)(l - 1) * FOURH;

    // xz = x @ W + b  (big WMMA GEMM, f32 out)
    wmma_gemm_bias<<<dim3(MROWS / 128, FOURH / 128), 256, 0, stream>>>(
        xcur, Wt, bias, xz, nullptr, MROWS, FOURH, Kin, 0);

    // h0 = c0 = 0
    fill_zero_u32<<<(B_ * H_ * 2 / 4 + 255) / 256, 256, 0, stream>>>(
        (unsigned int*)hping, B_ * H_ * 2 / 4);
    fill_zero_u32<<<(B_ * H_ + 255) / 256, 256, 0, stream>>>(
        (unsigned int*)cbuf, B_ * H_);

    // sequential scan, one launch per timestep (stream-ordered dependency)
    __bf16* hp = hping;
    __bf16* hq = hpong;
    for (int t = 0; t < S_; ++t) {
      lstm_step<<<dim3(B_ / 128, H_ / 32), 256, 0, stream>>>(
          xz, hp, Ut, cbuf, hq, xnext, t);
      __bf16* tmp = hp; hp = hq; hq = tmp;
    }
    __bf16* tx = xcur; xcur = xnext; xnext = tx;
  }

  // h1 = relu(x @ dw1 + db1)  -> bf16 (into freed xz region)
  wmma_gemm_bias<<<dim3(MROWS / 128, (2 * H_) / 128), 256, 0, stream>>>(
      xcur, dw1t, db1, nullptr, h1, MROWS, 2 * H_, H_, 1);

  // logits = h1 @ dw2 + db2  -> f32 d_out  (N=32, bounds-guarded tile)
  wmma_gemm_bias<<<dim3(MROWS / 128, 1), 256, 0, stream>>>(
      h1, dw2t, db2, out, nullptr, MROWS, V_, 2 * H_, 0);
}
